// MultiHeadedAttention_XL_1967095021842
// MI455X (gfx1250) — matrix operand order repaired and checked by reading, not compiled, not run on hardware
//
#include <hip/hip_runtime.h>
#include <cstdint>

// ---------------------------------------------------------------------------
// Transformer-XL multi-head attention for MI455X (gfx1250, wave32, WMMA).
// B=4, T=512, F=768, H=12, DK=64.
//
// Key optimization: never materialize r = r_emb @ Wr.T (805MB, 309 GFLOP).
// BD[b,h,t,l] = sum_f r_emb[t,l,f] * u[b,h,t,f],
//   u[b,h,t,f] = sum_d (q+bq+r_r_bias)[b,t,h,d] * Wr[h*64+d, f].
// The br contribution is constant over l -> softmax-shift-invariant -> dropped.
// ---------------------------------------------------------------------------

#define B_   4
#define T_   512
#define F_   768
#define H_   12
#define DK_  64
#define BH_  (B_ * H_)     // 48
#define NTOK (B_ * T_)     // 2048
#define MINVAL (-3.402823466e38f)

typedef __bf16 bf16;
typedef __attribute__((ext_vector_type(16))) __bf16 v16bf;
typedef __attribute__((ext_vector_type(8)))  float  v8f;
typedef __attribute__((ext_vector_type(4)))  unsigned int u32x4;
typedef __attribute__((ext_vector_type(8)))  int          i32x8;
typedef __attribute__((ext_vector_type(4)))  int          i32x4;

// ---------------- WMMA fragment loaders (contiguous k = 16*half + s, same for A and B) ----------------
// A operand 16x32 bf16: lane m(0-15)=row m : K=0..15
//                       lane m+16 =row m   : K=16..31
template <typename SrcT>
__device__ inline v16bf load_A(const SrcT* __restrict__ src, long ld) {
  const int lane = threadIdx.x & 31;
  const int half = lane >> 4, m = lane & 15;
  v16bf a;
#pragma unroll
  for (int j = 0; j < 8; ++j) {
    const int k = 2 * j + half * 16;
    a[2 * j]     = (bf16)(float)src[(long)m * ld + k];
    a[2 * j + 1] = (bf16)(float)src[(long)m * ld + k + 1];
  }
  return a;
}

// B operand 32x16 bf16: lane n(0-15)=col n, K=0..15 ; lane n+16=col n, K=16..31
// Source stored [N][K] row-major (B element (k,n) = src[n*ld+k]).
template <typename SrcT>
__device__ inline v16bf load_B_nk(const SrcT* __restrict__ src, long ld) {
  const int lane = threadIdx.x & 31;
  const int half = lane >> 4, n = lane & 15;
  v16bf b;
#pragma unroll
  for (int j = 0; j < 8; ++j) {
    const int k = 2 * j + half * 16;
    b[2 * j]     = (bf16)(float)src[(long)n * ld + k];
    b[2 * j + 1] = (bf16)(float)src[(long)n * ld + k + 1];
  }
  return b;
}

// Source stored [K][N] row-major (B element (k,n) = src[k*ld+n]).
template <typename SrcT>
__device__ inline v16bf load_B_kn(const SrcT* __restrict__ src, long ld) {
  const int lane = threadIdx.x & 31;
  const int half = lane >> 4, n = lane & 15;
  v16bf b;
#pragma unroll
  for (int j = 0; j < 8; ++j) {
    const int k = 2 * j + half * 16;
    b[2 * j]     = (bf16)(float)src[(long)k * ld + n];
    b[2 * j + 1] = (bf16)(float)src[(long)(k + 1) * ld + n];
  }
  return b;
}

__device__ inline v8f wmma_bf16(v16bf a, v16bf b, v8f c) {
  return __builtin_amdgcn_wmma_f32_16x16x32_bf16(false, a, false, b, (short)0, c,
                                                 false, false);
}

// ---------------------------- K0: weight -> bf16 ----------------------------
__global__ void convert_w(const float* __restrict__ src, bf16* __restrict__ dst,
                          int n) {
  int i = blockIdx.x * blockDim.x + threadIdx.x;
  if (i < n) dst[i] = (bf16)src[i];
}

// ------------------- K1: q/k/v projection, head-major out -------------------
// C[m,n] = sum_k X[m,k]*W[n,k] + bias[n]; out stored [BH, T, DK] bf16.
// dst1 gets +bias2a (may be null), dst2 (optional) gets +bias2b.
__global__ void __launch_bounds__(256)
proj_qkv(const float* __restrict__ X, const bf16* __restrict__ W,
         const float* __restrict__ bias, const float* __restrict__ bias2a,
         bf16* __restrict__ dst1, const float* __restrict__ bias2b,
         bf16* __restrict__ dst2) {
  const int w    = threadIdx.x >> 5;
  const int msub = w >> 1;
  const int nsub0 = (w & 1) * 2;
  const int m0 = blockIdx.x * 64 + msub * 16;
  v8f acc[2] = {};
  for (int k0 = 0; k0 < F_; k0 += 32) {
    v16bf a = load_A(X + (long)m0 * F_ + k0, F_);
#pragma unroll
    for (int j = 0; j < 2; ++j) {
      const int n0 = blockIdx.y * 64 + (nsub0 + j) * 16;
      v16bf b = load_B_nk(W + (long)n0 * F_ + k0, F_);
      acc[j] = wmma_bf16(a, b, acc[j]);
    }
  }
  const int lane = threadIdx.x & 31, half = lane >> 4, ln = lane & 15;
#pragma unroll
  for (int j = 0; j < 2; ++j) {
    const int col = blockIdx.y * 64 + (nsub0 + j) * 16 + ln;
    const int h = col >> 6, d = col & 63;
#pragma unroll
    for (int r = 0; r < 8; ++r) {
      const int m = m0 + r + half * 8;
      const int b = m >> 9, t = m & 511;
      const long di = (((long)(b * H_ + h)) * T_ + t) * DK_ + d;
      const float v = acc[j][r] + bias[col];
      dst1[di] = (bf16)(v + (bias2a ? bias2a[col] : 0.0f));
      if (dst2) dst2[di] = (bf16)(v + bias2b[col]);
    }
  }
}

// --------------------- K2: u = (q+bq+rrb) @ Wr_h (K=64) ---------------------
__global__ void __launch_bounds__(256)
compute_u(const bf16* __restrict__ qr, const bf16* __restrict__ Wr,
          bf16* __restrict__ u) {
  const int h = blockIdx.x;
  const int w = threadIdx.x >> 5;
  const int n0 = blockIdx.z * 128 + w * 16;
  v8f acc[4] = {};
#pragma unroll
  for (int ks = 0; ks < 2; ++ks) {
    const int k0 = ks * 32;
    v16bf b = load_B_kn(Wr + ((long)(h * DK_ + k0)) * F_ + n0, F_);
#pragma unroll
    for (int ms = 0; ms < 4; ++ms) {
      const int mrow = blockIdx.y * 64 + ms * 16;   // token index in [0,2048)
      const int bb = mrow >> 9, t0 = mrow & 511;
      v16bf a = load_A(qr + (((long)(bb * H_ + h)) * T_ + t0) * DK_ + k0, DK_);
      acc[ms] = wmma_bf16(a, b, acc[ms]);
    }
  }
  const int lane = threadIdx.x & 31, half = lane >> 4, ln = lane & 15;
#pragma unroll
  for (int ms = 0; ms < 4; ++ms) {
    const int mrow = blockIdx.y * 64 + ms * 16;
    const int bb = mrow >> 9, t0 = mrow & 511;
#pragma unroll
    for (int r = 0; r < 8; ++r) {
      const int t = t0 + r + half * 8;
      u[(((long)(bb * H_ + h)) * T_ + t) * F_ + n0 + ln] = (bf16)acc[ms][r];
    }
  }
}

// ------------- K3: BD[bh,t,l] = u[bh,t,:] . r_emb[t,l,:]  (TDM) -------------
// grid (t=512, lblk=4); block 256 = 8 waves; per block M=48, N=128, K=768.
// r_emb tiles (128 l x 32 f, f32, 16KB) staged into LDS by the Tensor Data
// Mover; compute waits on s_wait_tensorcnt.
__global__ void __launch_bounds__(256)
compute_bd(const float* __restrict__ r_emb, const bf16* __restrict__ u,
           float* __restrict__ bd) {
  __shared__ float tile[128 * 32];   // [l_local][f_local]
  const int t = blockIdx.x;
  const int lblk = blockIdx.y;
  const int w = threadIdx.x >> 5;
  v8f acc[3] = {};
  const long slab = (((long)t) * T_ + (long)lblk * 128) * F_;  // elem off of (t,l0,0)

  for (int f0 = 0; f0 < F_; f0 += 32) {
    __syncthreads();   // previous tile fully consumed
#if __has_builtin(__builtin_amdgcn_tensor_load_to_lds) && \
    __has_builtin(__builtin_amdgcn_s_wait_tensorcnt)
    if ((threadIdx.x >> 5) == 0) {
      const unsigned long long gaddr =
          (unsigned long long)(uintptr_t)r_emb + (unsigned long long)(slab + f0) * 4ull;
      const unsigned int ldsoff = (unsigned int)(uintptr_t)(void*)&tile[0];
      u32x4 g0;
      g0[0] = 1u;                                       // count=1, user mode
      g0[1] = ldsoff;                                   // lds_addr (bytes)
      g0[2] = (unsigned int)(gaddr & 0xffffffffull);    // global_addr[31:0]
      g0[3] = (unsigned int)((gaddr >> 32) & 0x1ffffffull) | (2u << 30); // type=2
      i32x8 g1;
      g1[0] = (2 << 16);                                // data_size=2 (4 bytes)
      g1[1] = (int)((F_ & 0xffff) << 16);               // tensor_dim0 lo16
      g1[2] = (int)(((F_ >> 16) & 0xffff) | ((T_ & 0xffff) << 16)); // dim0 hi|dim1 lo
      g1[3] = (int)(((T_ >> 16) & 0xffff) | (32 << 16));            // dim1 hi|tile_dim0=32
      g1[4] = 128;                                      // tile_dim1=128, tile_dim2=0
      g1[5] = F_;                                       // tensor_dim0_stride lo32
      g1[6] = (int)((((long)T_ * F_) & 0xffffl) << 16); // dim0_str hi|dim1_str lo16
      g1[7] = (int)(((long)T_ * F_) >> 16);             // dim1_stride[47:16]
      i32x4 gz4 = {};
      i32x8 gz8 = {};
      // clang-23 / therock-10.0 form: 6 args (g0, g1, g2, g3, ext, cpol)
      __builtin_amdgcn_tensor_load_to_lds(g0, g1, gz4, gz4, gz8, 0);
      __builtin_amdgcn_s_wait_tensorcnt(0);
    }
#else
    for (int i = threadIdx.x; i < 128 * 32; i += 256)
      tile[i] = r_emb[slab + (long)(i >> 5) * F_ + f0 + (i & 31)];
#endif
    __syncthreads();   // staged tile visible to all waves

    v16bf b = load_B_nk(&tile[(w * 16) * 32], 32);      // B[k=f][n=l] from LDS
#pragma unroll
    for (int ms = 0; ms < 3; ++ms) {
      const bf16* Ap = u + ((long)(ms * 16) * T_ + t) * F_ + f0;  // row stride T*F
      __builtin_prefetch(Ap + 32, 0, 0);                // global_prefetch_b8
      v16bf a = load_A(Ap, (long)T_ * F_);
      acc[ms] = wmma_bf16(a, b, acc[ms]);
    }
  }
  const int lane = threadIdx.x & 31, half = lane >> 4, ln = lane & 15;
  const int l = lblk * 128 + w * 16 + ln;
#pragma unroll
  for (int ms = 0; ms < 3; ++ms)
#pragma unroll
    for (int r = 0; r < 8; ++r) {
      const int bh = ms * 16 + r + half * 8;
      bd[((long)bh * T_ + t) * T_ + l] = acc[ms][r];
    }
}

// ----- K4: fused attention: AC (WMMA) + BD + mask + softmax + attn@V --------
// grid (bh=48, tb=16); block 256 = 8 waves handling 32 query rows x 512 keys.
__global__ void __launch_bounds__(256)
attention(const bf16* __restrict__ qw, const bf16* __restrict__ kb,
          const bf16* __restrict__ vb, const float* __restrict__ bd,
          const int* __restrict__ mask, float* __restrict__ xout) {
  const int bh = blockIdx.x, tb = blockIdx.y;
  const int b = bh / H_, h = bh % H_;
  const int w = threadIdx.x >> 5;
  const int lane = threadIdx.x & 31, half = lane >> 4, ln = lane & 15;
  __shared__ bf16 attn_s[32 * 512];     // 32KB normalized probs
  __shared__ float red[2][32 * 8];      // per-row partial max / sum (8 waves)

  // ---- scores = qw @ k^T ----
  v8f acc[2][4] = {};
#pragma unroll
  for (int ks = 0; ks < DK_; ks += 32) {
    v16bf a0 = load_A(qw + ((long)bh * T_ + tb * 32) * DK_ + ks, DK_);
    v16bf a1 = load_A(qw + ((long)bh * T_ + tb * 32 + 16) * DK_ + ks, DK_);
#pragma unroll
    for (int nj = 0; nj < 4; ++nj) {
      const int s0 = (w * 4 + nj) * 16;
      v16bf bf = load_B_nk(kb + ((long)bh * T_ + s0) * DK_ + ks, DK_);
      acc[0][nj] = wmma_bf16(a0, bf, acc[0][nj]);
      acc[1][nj] = wmma_bf16(a1, bf, acc[1][nj]);
    }
  }

  // ---- + BD, mask, scale; per-row partial max ----
  const float scale = 0.125f;  // 1/sqrt(64)
#pragma unroll
  for (int mi = 0; mi < 2; ++mi)
#pragma unroll
    for (int r = 0; r < 8; ++r) {
      const int tq = tb * 32 + mi * 16 + r + half * 8;
      float mx = MINVAL;
#pragma unroll
      for (int nj = 0; nj < 4; ++nj) {
        const int s = (w * 4 + nj) * 16 + ln;
        float v = (acc[mi][nj][r] + bd[((long)bh * T_ + tq) * T_ + s]) * scale;
        if (mask[((long)b * T_ + tq) * T_ + s] == 0) v = MINVAL;
        acc[mi][nj][r] = v;
        mx = fmaxf(mx, v);
      }
      for (int off = 1; off < 16; off <<= 1) mx = fmaxf(mx, __shfl_xor(mx, off, 32));
      if (ln == 0) red[0][(mi * 16 + r + half * 8) * 8 + w] = mx;
    }
  __syncthreads();

  // ---- exp(v - rowmax); per-row partial sum ----
#pragma unroll
  for (int mi = 0; mi < 2; ++mi)
#pragma unroll
    for (int r = 0; r < 8; ++r) {
      const int row = mi * 16 + r + half * 8;
      float rmax = red[0][row * 8 + 0];
#pragma unroll
      for (int q = 1; q < 8; ++q) rmax = fmaxf(rmax, red[0][row * 8 + q]);
      float sm = 0.0f;
#pragma unroll
      for (int nj = 0; nj < 4; ++nj) {
        float p = __expf(acc[mi][nj][r] - rmax);
        acc[mi][nj][r] = p;
        sm += p;
      }
      for (int off = 1; off < 16; off <<= 1) sm += __shfl_xor(sm, off, 32);
      if (ln == 0) red[1][row * 8 + w] = sm;
    }
  __syncthreads();

  // ---- normalize -> LDS bf16 ----
#pragma unroll
  for (int mi = 0; mi < 2; ++mi)
#pragma unroll
    for (int r = 0; r < 8; ++r) {
      const int row = mi * 16 + r + half * 8;
      float sm = 0.0f;
#pragma unroll
      for (int q = 0; q < 8; ++q) sm += red[1][row * 8 + q];
      const float inv = 1.0f / sm;
#pragma unroll
      for (int nj = 0; nj < 4; ++nj) {
        const int s = (w * 4 + nj) * 16 + ln;
        attn_s[row * 512 + s] = (bf16)(acc[mi][nj][r] * inv);
      }
    }
  __syncthreads();

  // ---- out = attn @ v : one 16x16 tile per wave ----
  const int mi2 = w >> 2, nj2 = w & 3;
  v8f o = {};
  for (int ks = 0; ks < T_; ks += 32) {
    v16bf a = load_A(&attn_s[(mi2 * 16) * 512 + ks], 512);
    v16bf bf = load_B_kn(vb + ((long)bh * T_ + ks) * DK_ + nj2 * 16, DK_);
    o = wmma_bf16(a, bf, o);
  }
#pragma unroll
  for (int r = 0; r < 8; ++r) {
    const int tq = tb * 32 + mi2 * 16 + r + half * 8;
    xout[((long)(b * T_ + tq)) * F_ + h * DK_ + nj2 * 16 + ln] = o[r];
  }
}

// -------------------------- K5: out = x @ Wo^T + bo -------------------------
__global__ void __launch_bounds__(256)
out_proj(const float* __restrict__ X, const bf16* __restrict__ Wo,
         const float* __restrict__ bo, float* __restrict__ out) {
  const int w = threadIdx.x >> 5;
  const int msub = w >> 1, nsub0 = (w & 1) * 2;
  const int m0 = blockIdx.x * 64 + msub * 16;
  v8f acc[2] = {};
  for (int k0 = 0; k0 < F_; k0 += 32) {
    v16bf a = load_A(X + (long)m0 * F_ + k0, F_);
#pragma unroll
    for (int j = 0; j < 2; ++j) {
      const int n0 = blockIdx.y * 64 + (nsub0 + j) * 16;
      v16bf b = load_B_nk(Wo + (long)n0 * F_ + k0, F_);
      acc[j] = wmma_bf16(a, b, acc[j]);
    }
  }
  const int lane = threadIdx.x & 31, half = lane >> 4, ln = lane & 15;
#pragma unroll
  for (int j = 0; j < 2; ++j) {
    const int col = blockIdx.y * 64 + (nsub0 + j) * 16 + ln;
#pragma unroll
    for (int r = 0; r < 8; ++r) {
      const int m = m0 + r + half * 8;
      out[(long)m * F_ + col] = acc[j][r] + bo[col];
    }
  }
}

// ------------------------------- launch ------------------------------------
extern "C" void kernel_launch(void* const* d_in, const int* in_sizes, int n_in,
                              void* d_out, int out_size, void* d_ws,
                              size_t ws_size, hipStream_t stream) {
  const float* query = (const float*)d_in[0];
  const float* key   = (const float*)d_in[1];
  const float* value = (const float*)d_in[2];
  const float* r_emb = (const float*)d_in[3];
  const int*   mask  = (const int*)d_in[4];
  const float* Wq = (const float*)d_in[5];  const float* bq = (const float*)d_in[6];
  const float* Wk = (const float*)d_in[7];  const float* bk = (const float*)d_in[8];
  const float* Wv = (const float*)d_in[9];  const float* bv = (const float*)d_in[10];
  const float* Wr = (const float*)d_in[11]; /* br (d_in[12]) softmax-invariant */
  const float* Wo = (const float*)d_in[13]; const float* bo = (const float*)d_in[14];
  const float* rwb = (const float*)d_in[15];  // (H,DK) flat = 768
  const float* rrb = (const float*)d_in[16];

  char* ws = (char*)d_ws;
  size_t off = 0;
  auto alloc = [&](size_t bytes) -> void* {
    void* p = ws + off;
    off = (off + bytes + 255) & ~(size_t)255;
    return p;
  };
  const size_t WN = (size_t)F_ * F_;                 // 589824
  bf16* Wq_b = (bf16*)alloc(WN * 2);
  bf16* Wk_b = (bf16*)alloc(WN * 2);
  bf16* Wv_b = (bf16*)alloc(WN * 2);
  bf16* Wr_b = (bf16*)alloc(WN * 2);
  bf16* Wo_b = (bf16*)alloc(WN * 2);
  const size_t QN = (size_t)BH_ * T_ * DK_;          // 1.57M
  bf16* qw = (bf16*)alloc(QN * 2);                   // q + bq + r_w_bias
  bf16* qr = (bf16*)alloc(QN * 2);                   // q + bq + r_r_bias
  bf16* kb = (bf16*)alloc(QN * 2);
  bf16* vb = (bf16*)alloc(QN * 2);
  bf16* ub = (bf16*)alloc((size_t)BH_ * T_ * F_ * 2);        // 37.7MB
  float* bdb = (float*)alloc((size_t)BH_ * T_ * T_ * 4);     // 50.3MB
  float* xb  = (float*)alloc((size_t)NTOK * F_ * 4);         // 6.3MB

  const dim3 blk(256);
  // K0: weights -> bf16
  {
    const int n = (int)WN, g = (n + 255) / 256;
    convert_w<<<g, blk, 0, stream>>>(Wq, Wq_b, n);
    convert_w<<<g, blk, 0, stream>>>(Wk, Wk_b, n);
    convert_w<<<g, blk, 0, stream>>>(Wv, Wv_b, n);
    convert_w<<<g, blk, 0, stream>>>(Wr, Wr_b, n);
    convert_w<<<g, blk, 0, stream>>>(Wo, Wo_b, n);
  }
  // K1: projections
  {
    dim3 grid(NTOK / 64, F_ / 64);
    proj_qkv<<<grid, blk, 0, stream>>>(query, Wq_b, bq, rwb, qw, rrb, qr);
    proj_qkv<<<grid, blk, 0, stream>>>(key, Wk_b, bk, nullptr, kb, nullptr, nullptr);
    proj_qkv<<<grid, blk, 0, stream>>>(value, Wv_b, bv, nullptr, vb, nullptr, nullptr);
  }
  // K2: u
  compute_u<<<dim3(H_, NTOK / 64, F_ / 128), blk, 0, stream>>>(qr, Wr_b, ub);
  // K3: BD (TDM-staged r_emb stream — the bandwidth-critical kernel)
  compute_bd<<<dim3(T_, T_ / 128), blk, 0, stream>>>(r_emb, ub, bdb);
  // K4: attention
  attention<<<dim3(BH_, T_ / 32), blk, 0, stream>>>(qw, kb, vb, bdb, mask, xb);
  // K5: output projection
  out_proj<<<dim3(NTOK / 64, F_ / 64), blk, 0, stream>>>(xb, Wo_b, bo,
                                                         (float*)d_out);
}
